// Multi_Head_Self_Attention_76768245448826
// MI455X (gfx1250) — matrix operand order repaired and checked by reading, not compiled, not run on hardware
//
#include <hip/hip_runtime.h>

// ---------------- constants ----------------
#define BB   8
#define MM   1024
#define HID  512
#define HH   8
#define DD   64
#define ROWS (BB*MM)          // 8192
#define NNEI 32

typedef __attribute__((ext_vector_type(8)))  __bf16 v8bf;
typedef __attribute__((ext_vector_type(16))) __bf16 v16bf;
typedef __attribute__((ext_vector_type(8)))  float  v8f;
typedef __attribute__((ext_vector_type(4)))  int    v4i;

#if __has_builtin(__builtin_amdgcn_global_load_async_to_lds_b128)
#define HAVE_ASYNC_LDS 1
typedef __attribute__((address_space(1))) v4i* as1_v4i;
typedef __attribute__((address_space(3))) v4i* as3_v4i;
#endif

// ---------------- helpers ----------------
static __device__ __forceinline__ unsigned short f2bfu(float f) {
    __bf16 h = (__bf16)f;                       // native v_cvt on gfx1250
    return __builtin_bit_cast(unsigned short, h);
}
static __device__ __forceinline__ v16bf cat8(v8bf lo, v8bf hi) {
    return __builtin_shufflevector(lo, hi, 0,1,2,3,4,5,6,7,8,9,10,11,12,13,14,15);
}
static __device__ __forceinline__ v8bf ld8(const unsigned short* p) {
    return *(const v8bf*)p;                     // 16B load (global_load_b128 / ds_load_b128)
}
static __device__ __forceinline__ float mishf(float x) {
    float sp = (x > 20.0f) ? x : log1pf(__expf(x));
    return x * tanhf(sp);
}

// =====================================================================
// 0a) f32 -> bf16 bulk convert (row-major), vectorized 4-wide
// =====================================================================
__global__ __launch_bounds__(256) void cvt_x_kernel(
    const float* __restrict__ in, unsigned short* __restrict__ out)
{
    int idx = blockIdx.x * 256 + threadIdx.x;   // n/4 threads
    float4 v = ((const float4*)in)[idx];
    ushort4 o;
    o.x = f2bfu(v.x); o.y = f2bfu(v.y); o.z = f2bfu(v.z); o.w = f2bfu(v.w);
    ((ushort4*)out)[idx] = o;
}

// =====================================================================
// 0b) f32 W[k][n] -> bf16 Wt[n][k] (transposed, so GEMM B-loads are contiguous)
// =====================================================================
__global__ __launch_bounds__(256) void cvt_w_t_kernel(
    const float* __restrict__ W, unsigned short* __restrict__ Wt)
{
    int idx = blockIdx.x * 256 + threadIdx.x;   // 512*512 threads
    int n = idx >> 9, k = idx & 511;
    Wt[idx] = f2bfu(W[(size_t)k * HID + n]);
}

// =====================================================================
// 1) GEMM: acc = Xb[8192x512] * Wt^T (+bias). MODE: 0 = bf16 row-major out,
//    1 = bf16 transposed out (for V: Vt[(b*512+col)*1024 + m]),
//    2 = f32 out with Mish (final projection).
// =====================================================================
template <int MODE, typename OutT>
__global__ __launch_bounds__(128) void gemm_bf16(
    const unsigned short* __restrict__ Xb, const unsigned short* __restrict__ Wt,
    const float* __restrict__ bias, OutT* __restrict__ Out)
{
    const int lane = threadIdx.x & 31;
    const int wv   = threadIdx.x >> 5;
    const int w    = blockIdx.x * 4 + wv;           // 4096 waves
    const int rowTile = w >> 3;
    const int colTile = w & 7;
    const int half = lane >> 4;
    const int l16  = lane & 15;
    const int r    = rowTile * 16 + l16;

    v8f acc[4];
#pragma unroll
    for (int n = 0; n < 4; ++n) acc[n] = (v8f){0.f,0.f,0.f,0.f,0.f,0.f,0.f,0.f};

    for (int kc = 0; kc < HID / 32; ++kc) {
        const unsigned short* ap = Xb + (size_t)r * HID + kc * 32;
        v16bf a = cat8(ld8(ap + 8 * half), ld8(ap + 16 + 8 * half));
#pragma unroll
        for (int n = 0; n < 4; ++n) {
            int col = colTile * 64 + n * 16 + l16;
            const unsigned short* bp = Wt + (size_t)col * HID + kc * 32 + 8 * half;
            v16bf bfr = cat8(ld8(bp), ld8(bp + 16));
            acc[n] = __builtin_amdgcn_wmma_f32_16x16x32_bf16(
                false, a, false, bfr, (short)0, acc[n], false, false);
        }
    }
#pragma unroll
    for (int n = 0; n < 4; ++n) {
        int col  = colTile * 64 + n * 16 + l16;
        float bb = bias[col];
#pragma unroll
        for (int v = 0; v < 8; ++v) {
            int m = rowTile * 16 + v + 8 * half;
            float val = acc[n][v] + bb;
            if constexpr (MODE == 0) {
                Out[(size_t)m * HID + col] = f2bfu(val);
            } else if constexpr (MODE == 1) {
                int bcur = m >> 10, mm = m & (MM - 1);
                Out[((size_t)bcur * HID + col) * MM + mm] = f2bfu(val);
            } else {
                Out[(size_t)m * HID + col] = mishf(val);
            }
        }
    }
}

// =====================================================================
// 2) ew = e * (rank < 32) / (rowsum + 1e-5); stable descending rank.
// =====================================================================
__global__ __launch_bounds__(256) void ew_kernel(
    const float* __restrict__ e, float* __restrict__ ew)
{
    __shared__ float srow[MM];
    __shared__ float ssum[256];
    const int tid = threadIdx.x;
    const size_t row = blockIdx.x;
    const float* ep = e + row * MM;

    for (int i = tid; i < MM; i += 256) srow[i] = ep[i];
    __syncthreads();

    float kept[4];
#pragma unroll
    for (int t = 0; t < 4; ++t) {
        int idx = tid + t * 256;
        float v = srow[idx];
        int cnt = 0;
        for (int j = 0; j < MM; j += 4) {
            float a0 = srow[j], a1 = srow[j+1], a2 = srow[j+2], a3 = srow[j+3];
            cnt += (a0 > v) || (a0 == v && (j    ) < idx);
            cnt += (a1 > v) || (a1 == v && (j + 1) < idx);
            cnt += (a2 > v) || (a2 == v && (j + 2) < idx);
            cnt += (a3 > v) || (a3 == v && (j + 3) < idx);
        }
        kept[t] = (cnt < NNEI) ? v : 0.0f;
    }
    ssum[tid] = kept[0] + kept[1] + kept[2] + kept[3];
    __syncthreads();
    for (int s2 = 128; s2 > 0; s2 >>= 1) {
        if (tid < s2) ssum[tid] += ssum[tid + s2];
        __syncthreads();
    }
    float inv = 1.0f / (ssum[0] + 1e-5f);
#pragma unroll
    for (int t = 0; t < 4; ++t)
        ew[row * MM + tid + t * 256] = kept[t] * inv;
}

// =====================================================================
// 3) Flash attention. Block = one (b,h,64-query group), 4 waves = 4 query tiles of 16.
//    K tile (32x64) and Vt tile (64x32) staged into LDS once per key tile
//    (async global->LDS when available), shared by all 4 waves.
// =====================================================================
__global__ __launch_bounds__(128) void attn_kernel(
    const unsigned short* __restrict__ Q, const unsigned short* __restrict__ K,
    const unsigned short* __restrict__ Vt, const float* __restrict__ ew,
    const int* __restrict__ masks, unsigned short* __restrict__ AOut)
{
    __shared__ __align__(16) unsigned short kTile[32 * 64];   // [key][d]
    __shared__ __align__(16) unsigned short vTile[64 * 32];   // [d][key]
    __shared__ __align__(16) unsigned short pshare[4][16 * 32];

    const int tid  = threadIdx.x;
    const int lane = tid & 31;
    const int wv   = tid >> 5;
    const int b    = blockIdx.x / (HH * (MM / 64));
    const int rem  = blockIdx.x % (HH * (MM / 64));
    const int h    = rem / (MM / 64);
    const int qg   = rem % (MM / 64);
    const int qt   = qg * 4 + wv;
    const int half = lane >> 4;
    const int l16  = lane & 15;
    const int qbase = qt * 16;

    const size_t rowbase = (size_t)b * MM;
    const int    hcol    = h * DD;

    // Q A-fragments (16x64, two K-chunks), loaded once as 4x16B
    v16bf qa[2];
    {
        const unsigned short* qp = Q + (rowbase + qbase + l16) * HID + hcol;
#pragma unroll
        for (int c = 0; c < 2; ++c)
            qa[c] = cat8(ld8(qp + c * 32 + 8 * half), ld8(qp + c * 32 + 16 + 8 * half));
    }

    v8f o[4];
#pragma unroll
    for (int n = 0; n < 4; ++n) o[n] = (v8f){0.f,0.f,0.f,0.f,0.f,0.f,0.f,0.f};
    float mr[8], lr[8];
#pragma unroll
    for (int v = 0; v < 8; ++v) { mr[v] = -3.0e38f; lr[v] = 0.0f; }

    for (int kt = 0; kt < MM / 32; ++kt) {
        const int keybase = kt * 32;
        const unsigned short* Kp = K  + (rowbase + keybase) * HID + hcol;        // row stride HID
        const unsigned short* Vp = Vt + ((size_t)b * HID + hcol) * MM + keybase; // row stride MM

        // ---- stage K (32 rows x 128B) and Vt (64 rows x 64B) into LDS ----
        for (int s = tid; s < 256; s += 128) {
            int kr = s >> 3, ko = (s & 7) * 8;      // K: 8 x 16B per row
            int vr = s >> 2, vo = (s & 3) * 8;      // V: 4 x 16B per row
#ifdef HAVE_ASYNC_LDS
            __builtin_amdgcn_global_load_async_to_lds_b128(
                (as1_v4i)(Kp + (size_t)kr * HID + ko),
                (as3_v4i)(kTile + kr * 64 + ko), 0, 0);
            __builtin_amdgcn_global_load_async_to_lds_b128(
                (as1_v4i)(Vp + (size_t)vr * MM + vo),
                (as3_v4i)(vTile + vr * 32 + vo), 0, 0);
#else
            *(v8bf*)(kTile + kr * 64 + ko) = ld8(Kp + (size_t)kr * HID + ko);
            *(v8bf*)(vTile + vr * 32 + vo) = ld8(Vp + (size_t)vr * MM + vo);
#endif
        }
#ifdef HAVE_ASYNC_LDS
#if __has_builtin(__builtin_amdgcn_s_wait_asynccnt)
        __builtin_amdgcn_s_wait_asynccnt(0);
#else
        asm volatile("s_wait_asynccnt 0x0" ::: "memory");
#endif
#endif
        __syncthreads();

        // ---- scores: Q (16x64) x K^T -> two 16x16 tiles, from LDS ----
        v8f s0 = (v8f){0.f,0.f,0.f,0.f,0.f,0.f,0.f,0.f};
        v8f s1 = s0;
#pragma unroll
        for (int c = 0; c < 2; ++c) {
            const unsigned short* k0 = kTile + l16 * 64 + c * 32 + 8 * half;
            const unsigned short* k1 = kTile + (16 + l16) * 64 + c * 32 + 8 * half;
            v16bf bk0 = cat8(ld8(k0), ld8(k0 + 16));
            v16bf bk1 = cat8(ld8(k1), ld8(k1 + 16));
            s0 = __builtin_amdgcn_wmma_f32_16x16x32_bf16(false, qa[c], false, bk0, (short)0, s0, false, false);
            s1 = __builtin_amdgcn_wmma_f32_16x16x32_bf16(false, qa[c], false, bk1, (short)0, s1, false, false);
        }

        // ---- mask + ew scale + online softmax ----
        const int key0 = keybase + l16;
        const int key1 = keybase + 16 + l16;
        const int mk0  = masks[b * MM + key0];
        const int mk1  = masks[b * MM + key1];

#pragma unroll
        for (int v = 0; v < 8; ++v) {
            int m = qbase + v + 8 * half;
            size_t eoff = ((size_t)(b * MM + m)) * MM;
            float e0 = ew[eoff + key0];
            float e1 = ew[eoff + key1];
            float sv0 = (mk0 ? s0[v] : -1.0e12f) * e0 * 0.125f;
            float sv1 = (mk1 ? s1[v] : -1.0e12f) * e1 * 0.125f;

            float mx = fmaxf(sv0, sv1);
#pragma unroll
            for (int off = 1; off <= 8; off <<= 1)
                mx = fmaxf(mx, __shfl_xor(mx, off, 32));
            float mnew  = fmaxf(mr[v], mx);
            float alpha = __expf(mr[v] - mnew);
            float p0 = __expf(sv0 - mnew);
            float p1 = __expf(sv1 - mnew);
            float ps = p0 + p1;
#pragma unroll
            for (int off = 1; off <= 8; off <<= 1)
                ps += __shfl_xor(ps, off, 32);
            lr[v] = lr[v] * alpha + ps;
            mr[v] = mnew;
#pragma unroll
            for (int n = 0; n < 4; ++n) o[n][v] *= alpha;

            pshare[wv][(v + 8 * half) * 32 + l16]      = f2bfu(p0);
            pshare[wv][(v + 8 * half) * 32 + 16 + l16] = f2bfu(p1);
        }
        __syncthreads();

        // ---- P as A-fragment from LDS, O += P x V (from LDS) ----
        const unsigned short* pp = pshare[wv] + l16 * 32;
        v16bf pa = cat8(ld8(pp + 8 * half), ld8(pp + 16 + 8 * half));
#pragma unroll
        for (int dc = 0; dc < 4; ++dc) {
            const unsigned short* vp = vTile + (dc * 16 + l16) * 32 + 8 * half;
            v16bf bv = cat8(ld8(vp), ld8(vp + 16));
            o[dc] = __builtin_amdgcn_wmma_f32_16x16x32_bf16(false, pa, false, bv, (short)0, o[dc], false, false);
        }
        __syncthreads();   // protect kTile/vTile/pshare before next stage
    }

    // ---- normalize and store (b, m, h*64 + d) as bf16 ----
#pragma unroll
    for (int v = 0; v < 8; ++v) {
        float inv = 1.0f / (lr[v] + 1e-20f);
        int m = qbase + v + 8 * half;
#pragma unroll
        for (int dc = 0; dc < 4; ++dc) {
            int col = hcol + dc * 16 + l16;
            AOut[(rowbase + m) * HID + col] = f2bfu(o[dc][v] * inv);
        }
    }
}

// =====================================================================
// host launcher
// =====================================================================
extern "C" void kernel_launch(void* const* d_in, const int* in_sizes, int n_in,
                              void* d_out, int out_size, void* d_ws, size_t ws_size,
                              hipStream_t stream) {
    (void)in_sizes; (void)n_in; (void)out_size; (void)ws_size;
    const float* x     = (const float*)d_in[0];
    const float* e     = (const float*)d_in[1];
    const int*   masks = (const int*)  d_in[2];
    const float* Wq = (const float*)d_in[3];  const float* bq = (const float*)d_in[4];
    const float* Wk = (const float*)d_in[5];  const float* bk = (const float*)d_in[6];
    const float* Wv = (const float*)d_in[7];  const float* bv = (const float*)d_in[8];
    const float* Wo = (const float*)d_in[9];  const float* bo = (const float*)d_in[10];
    float* out = (float*)d_out;

    char* ws = (char*)d_ws;
    unsigned short* Qb  = (unsigned short*)(ws + ((size_t)0));
    unsigned short* Kb  = (unsigned short*)(ws + ((size_t)8  << 20));
    unsigned short* Vtb = (unsigned short*)(ws + ((size_t)16 << 20));
    float*          ewb = (float*)         (ws + ((size_t)24 << 20));
    unsigned short* Ab  = (unsigned short*)(ws + ((size_t)56 << 20));
    unsigned short* Xb  = (unsigned short*)(ws + ((size_t)64 << 20));
    unsigned short* Wtq = (unsigned short*)(ws + ((size_t)72 << 20));
    unsigned short* Wtk = Wtq + (size_t)HID * HID;
    unsigned short* Wtv = Wtk + (size_t)HID * HID;
    unsigned short* Wto = Wtv + (size_t)HID * HID;

    // 0) convert inputs to bf16 (X row-major, W transposed)
    cvt_x_kernel<<<dim3((ROWS * HID) / 4 / 256), dim3(256), 0, stream>>>(x, Xb);
    cvt_w_t_kernel<<<dim3((HID * HID) / 256), dim3(256), 0, stream>>>(Wq, Wtq);
    cvt_w_t_kernel<<<dim3((HID * HID) / 256), dim3(256), 0, stream>>>(Wk, Wtk);
    cvt_w_t_kernel<<<dim3((HID * HID) / 256), dim3(256), 0, stream>>>(Wv, Wtv);
    cvt_w_t_kernel<<<dim3((HID * HID) / 256), dim3(256), 0, stream>>>(Wo, Wto);

    dim3 gBlk(128);
    dim3 gGrd(1024);   // 4096 waves

    // 1) projections
    gemm_bf16<0, unsigned short><<<gGrd, gBlk, 0, stream>>>(Xb, Wtq, bq, Qb);
    gemm_bf16<0, unsigned short><<<gGrd, gBlk, 0, stream>>>(Xb, Wtk, bk, Kb);
    gemm_bf16<1, unsigned short><<<gGrd, gBlk, 0, stream>>>(Xb, Wtv, bv, Vtb); // V transposed

    // 2) edge weights
    ew_kernel<<<dim3(ROWS), dim3(256), 0, stream>>>(e, ewb);

    // 3) flash attention
    attn_kernel<<<gGrd, gBlk, 0, stream>>>(Qb, Kb, Vtb, ewb, masks, Ab);

    // 4) output projection + Mish
    gemm_bf16<2, float><<<gGrd, gBlk, 0, stream>>>(Ab, Wto, bo, out);
}
